// NonLocalBlock1D_54107997995619
// MI455X (gfx1250) — compile-verified, matrix operand order checked by reading, not run on hardware
//
#include <hip/hip_runtime.h>

// ---------------------------------------------------------------------------
// NonLocalBlock1D for MI455X (gfx1250, wave32, WMMA bf16 16x16x32, f32 acc)
// B=4, C=N=4096, INTER=64.
// All operand panels are pre-converted to bf16 so GEMM inner loops are pure
// {global_load_b128 x2 per fragment} + v_wmma.  sched_group_barrier pins a
// "all VMEM reads, then all WMMAs" order per K-step so fragment buffers get
// distinct registers and loads pipeline ahead of the XDL pipe.
// f [4096x4096] is never materialized (online column softmax + fused recompute).
// ---------------------------------------------------------------------------

#define NN 4096
#define NI 64
#define NB 4

// sched_group_barrier masks
#define SG_MFMA      0x008   // MFMA/WMMA
#define SG_VMEM_READ 0x020

typedef __attribute__((ext_vector_type(16))) __bf16 v16bf;
typedef __attribute__((ext_vector_type(4)))  __bf16 v4bf;
typedef __attribute__((ext_vector_type(8)))  float  v8f;

__device__ __forceinline__ __bf16 f2bf(float x) { return (__bf16)x; }

// bf16 fragment load: element h -> K = (h/8)*16 + kA + (h&7); the two groups
// of 8 bf16 are 16B-contiguous -> two global_load_b128 per fragment.
__device__ __forceinline__ v16bf load_frag_bf(const __bf16* __restrict__ p) {
  v16bf f;
  ((uint4*)&f)[0] = *(const uint4*)(p);
  ((uint4*)&f)[1] = *(const uint4*)(p + 16);
  return f;
}

// f32 fragment load with on-the-fly conversion (used only for x in k_proj).
__device__ __forceinline__ v16bf load_frag_f32(const float* __restrict__ p) {
  float4 a0 = *(const float4*)(p);
  float4 a1 = *(const float4*)(p + 4);
  float4 b0 = *(const float4*)(p + 16);
  float4 b1 = *(const float4*)(p + 20);
  v16bf f;
  f[0]=f2bf(a0.x);  f[1]=f2bf(a0.y);  f[2]=f2bf(a0.z);  f[3]=f2bf(a0.w);
  f[4]=f2bf(a1.x);  f[5]=f2bf(a1.y);  f[6]=f2bf(a1.z);  f[7]=f2bf(a1.w);
  f[8]=f2bf(b0.x);  f[9]=f2bf(b0.y);  f[10]=f2bf(b0.z); f[11]=f2bf(b0.w);
  f[12]=f2bf(b1.x); f[13]=f2bf(b1.y); f[14]=f2bf(b1.z); f[15]=f2bf(b1.w);
  return f;
}

__device__ __forceinline__ v8f wmma_bf16(v16bf a, v16bf b, v8f c) {
  return __builtin_amdgcn_wmma_f32_16x16x32_bf16(false, a, false, b,
                                                 (short)0, c, false, false);
}

__device__ __forceinline__ v8f vzero8() {
  v8f z = {0.f,0.f,0.f,0.f,0.f,0.f,0.f,0.f};
  return z;
}

// ---------------------------------------------------------------------------
// Kernel 0: f32 -> bf16 panel conversion (weights).
// ---------------------------------------------------------------------------
__global__ __launch_bounds__(256) void k_cvt(const float* __restrict__ s,
                                             __bf16* __restrict__ d) {
  const int i = blockIdx.x * 256 + threadIdx.x;     // group of 4
  float4 v = ((const float4*)s)[i];
  v4bf o;
  o[0] = f2bf(v.x); o[1] = f2bf(v.y); o[2] = f2bf(v.z); o[3] = f2bf(v.w);
  ((v4bf*)d)[i] = o;
}

// ---------------------------------------------------------------------------
// Kernel 1: projections.  P[l,o] = sum_k x[b,l,k]*w[o,k] + bias[o].
// One wave per block; 16 l-rows x 64 o x 3 matrices per block (x read once).
// Scheduling: all 28 VMEM reads issued first, then the 12 WMMAs.
// ---------------------------------------------------------------------------
__global__ __launch_bounds__(32, 1) void k_proj(
    const float* __restrict__ x,
    const __bf16* __restrict__ gw, const float* __restrict__ gb,
    const __bf16* __restrict__ tw, const float* __restrict__ tb,
    const __bf16* __restrict__ pw, const float* __restrict__ pb,
    float* __restrict__ Pg, __bf16* __restrict__ PthB, __bf16* __restrict__ PphB)
{
  const int lane = threadIdx.x;
  const int b    = blockIdx.y;
  const int l0   = blockIdx.x * 16;
  const int row  = lane & 15;
  const int half = lane >> 4;
  const int kA   = half * 8;

  const float* xrow = x + ((size_t)b * NN + (l0 + row)) * NN + kA;

  // 12 logical B tiles: j = p*4 + t; base pointer per p, t folds into the
  // instruction offset (t*16*NN elements).
  const __bf16* wbase[3];
  wbase[0] = gw + (size_t)row * NN + kA;
  wbase[1] = tw + (size_t)row * NN + kA;
  wbase[2] = pw + (size_t)row * NN + kA;

  v8f acc[12];
#pragma unroll
  for (int j = 0; j < 12; ++j) acc[j] = vzero8();

  for (int k0 = 0; k0 < NN; k0 += 32) {
    v16bf a = load_frag_f32(xrow + k0);
    v16bf bfr[12];
#pragma unroll
    for (int j = 0; j < 12; ++j) {
      const int p = j >> 2, t = j & 3;
      bfr[j] = load_frag_bf(wbase[p] + (size_t)(t * 16) * NN + k0);
    }
#pragma unroll
    for (int j = 0; j < 12; ++j)
      acc[j] = wmma_bf16(a, bfr[j], acc[j]);
    // pin schedule: 28 VMEM reads (4 A + 24 B) first, then 12 WMMAs
    __builtin_amdgcn_sched_group_barrier(SG_VMEM_READ, 28, 0);
    __builtin_amdgcn_sched_group_barrier(SG_MFMA,      12, 0);
  }

  const float* bias[3] = { gb, tb, pb };
#pragma unroll
  for (int t = 0; t < 4; ++t) {
    const int o = t * 16 + row;                // C-layout column
    const float big = bias[0][o], bit = bias[1][o], bip = bias[2][o];
#pragma unroll
    for (int v = 0; v < 8; ++v) {
      const int l = l0 + v + 8 * half;         // C-layout row
      const size_t idx = ((size_t)b * NN + l) * NI + o;
      Pg[idx]   = acc[0 * 4 + t][v] + big;
      PthB[idx] = f2bf(acc[1 * 4 + t][v] + bit);
      PphB[idx] = f2bf(acc[2 * 4 + t][v] + bip);
    }
  }
}

// ---------------------------------------------------------------------------
// Kernel 2: column softmax stats via online max/sum; f recomputed with WMMA.
// Lane pair (L, L+16) merged with shfl_xor(16) (wave32).
// ---------------------------------------------------------------------------
__global__ __launch_bounds__(32, 1) void k_stats(
    const __bf16* __restrict__ PthB, const __bf16* __restrict__ PphB,
    float* __restrict__ cmax, float* __restrict__ csum)
{
  const int lane = threadIdx.x;
  const int b    = blockIdx.y;
  const int m0   = blockIdx.x * 16;
  const int row  = lane & 15;
  const int half = lane >> 4;
  const int kA   = half * 8;

  const __bf16* phrow = PphB + ((size_t)b * NN + m0 + row) * NI + kA;
  const v16bf bf0 = load_frag_bf(phrow);        // i = 0..31
  const v16bf bf1 = load_frag_bf(phrow + 32);   // i = 32..63

  const __bf16* tha = PthB + ((size_t)b * NN + row) * NI + kA;

  float mrun = -3.0e38f, srun = 0.f;
  for (int n0 = 0; n0 < NN; n0 += 16) {
    const __bf16* ta = tha + (size_t)n0 * NI;
    const v16bf a0 = load_frag_bf(ta);
    const v16bf a1 = load_frag_bf(ta + 32);
    v8f acc = vzero8();
    acc = wmma_bf16(a0, bf0, acc);
    acc = wmma_bf16(a1, bf1, acc);
    float tmax = acc[0];
#pragma unroll
    for (int v = 1; v < 8; ++v) tmax = fmaxf(tmax, acc[v]);
    const float mnew = fmaxf(mrun, tmax);
    float s = srun * __expf(mrun - mnew);
#pragma unroll
    for (int v = 0; v < 8; ++v) s += __expf(acc[v] - mnew);
    srun = s; mrun = mnew;
    __builtin_amdgcn_sched_group_barrier(SG_VMEM_READ, 4, 0);
    __builtin_amdgcn_sched_group_barrier(SG_MFMA,      2, 0);
  }
  const float om = __shfl_xor(mrun, 16);
  const float os = __shfl_xor(srun, 16);
  const float mm = fmaxf(mrun, om);
  const float ss = srun * __expf(mrun - mm) + os * __expf(om - mm);
  if (lane < 16) {
    cmax[(size_t)b * NN + m0 + lane] = mm;
    csum[(size_t)b * NN + m0 + lane] = ss;
  }
}

// ---------------------------------------------------------------------------
// Kernel 3: gsTB[b][i][m] = bf16( P_g[b][m][i] / csum[b][m] ).
// ---------------------------------------------------------------------------
__global__ __launch_bounds__(256) void k_scale_g(
    const float* __restrict__ Pg, const float* __restrict__ csum,
    __bf16* __restrict__ gsTB)
{
  const int m = blockIdx.x * 256 + threadIdx.x;
  const int i = blockIdx.y;
  const int b = blockIdx.z;
  const float inv = 1.0f / csum[(size_t)b * NN + m];
  gsTB[((size_t)b * NI + i) * NN + m] =
      f2bf(Pg[((size_t)b * NN + m) * NI + i] * inv);
}

// ---------------------------------------------------------------------------
// Kernel 4: y[n,i] = sum_m exp(f[n,m]-cmax[m]) * gsT[i,m].
// Per m-chunk of 32: 4 WMMAs recompute f, exp in-register, restage the exp
// tile through LDS (C-layout f32 -> A-layout bf16), 4 WMMAs accumulate y.
// ---------------------------------------------------------------------------
__global__ __launch_bounds__(32, 1) void k_attn(
    const __bf16* __restrict__ PthB, const __bf16* __restrict__ PphB,
    const __bf16* __restrict__ gsTB, const float* __restrict__ cmax,
    __bf16* __restrict__ YB)
{
  __shared__ __bf16 elds[16 * 40];             // 16 n-rows x 32 m (+8 pad)

  const int lane = threadIdx.x;
  const int b    = blockIdx.y;
  const int n0   = blockIdx.x * 16;
  const int row  = lane & 15;
  const int half = lane >> 4;
  const int kA   = half * 8;

  const __bf16* tha = PthB + ((size_t)b * NN + n0 + row) * NI + kA;
  const v16bf a_th0 = load_frag_bf(tha);
  const v16bf a_th1 = load_frag_bf(tha + 32);

  const __bf16* gbase = gsTB + ((size_t)b * NI + row) * NN + kA; // + (t*16)*NN
  const __bf16* phbase = PphB + ((size_t)b * NN + row) * NI + kA;
  const float*  cmb    = cmax + (size_t)b * NN;

  v8f yacc[4];
#pragma unroll
  for (int t = 0; t < 4; ++t) yacc[t] = vzero8();

  for (int m0 = 0; m0 < NN; m0 += 32) {
    // batch all global reads for this m-chunk: phi (8), g' (8), cmax (2)
    const __bf16* pr0 = phbase + (size_t)m0 * NI;
    const __bf16* pr1 = phbase + (size_t)(m0 + 16) * NI;
    const v16bf p00 = load_frag_bf(pr0);
    const v16bf p01 = load_frag_bf(pr0 + 32);
    const v16bf p10 = load_frag_bf(pr1);
    const v16bf p11 = load_frag_bf(pr1 + 32);
    v16bf gf[4];
#pragma unroll
    for (int t = 0; t < 4; ++t)
      gf[t] = load_frag_bf(gbase + (size_t)(t * 16) * NN + m0);
    const float cm0 = cmb[m0 + row];
    const float cm1 = cmb[m0 + 16 + row];

    v8f e0 = vzero8(), e1 = vzero8();
    e0 = wmma_bf16(a_th0, p00, e0);
    e0 = wmma_bf16(a_th1, p01, e0);
    e1 = wmma_bf16(a_th0, p10, e1);
    e1 = wmma_bf16(a_th1, p11, e1);

#pragma unroll
    for (int v = 0; v < 8; ++v) e0[v] = __expf(e0[v] - cm0);
#pragma unroll
    for (int v = 0; v < 8; ++v) e1[v] = __expf(e1[v] - cm1);

    // stage E[n][m_local] as bf16: n = v+8*half, m_local = row (+16)
#pragma unroll
    for (int v = 0; v < 8; ++v) {
      elds[(v + 8 * half) * 40 + row]      = f2bf(e0[v]);
      elds[(v + 8 * half) * 40 + row + 16] = f2bf(e1[v]);
    }
    asm volatile("s_wait_dscnt 0" ::: "memory");

    // reload E as an A-layout 16x32 bf16 fragment
    v16bf ea;
    {
      const __bf16* ep = elds + row * 40 + kA;
      ((uint4*)&ea)[0] = *(const uint4*)(ep);
      ((uint4*)&ea)[1] = *(const uint4*)(ep + 16);
    }
#pragma unroll
    for (int t = 0; t < 4; ++t)
      yacc[t] = wmma_bf16(ea, gf[t], yacc[t]);

    // pin: 18 global reads first, then the 4 f-WMMAs (rest is dep-ordered)
    __builtin_amdgcn_sched_group_barrier(SG_VMEM_READ, 18, 0);
    __builtin_amdgcn_sched_group_barrier(SG_MFMA,       4, 0);
  }

#pragma unroll
  for (int t = 0; t < 4; ++t)
#pragma unroll
    for (int v = 0; v < 8; ++v)
      YB[((size_t)b * NN + n0 + v + 8 * half) * NI + t * 16 + row] =
          f2bf(yacc[t][v]);
}

// ---------------------------------------------------------------------------
// Kernel 5: out[b,o,l] = sum_i y[l,i]*W_w[o,i] + W_b[o] + x[b,l,o].
// One wave per (b, l-tile, o-chunk); vectorized float4 stores along l.
// ---------------------------------------------------------------------------
__global__ __launch_bounds__(32, 1) void k_out(
    const __bf16* __restrict__ YB,
    const __bf16* __restrict__ WwB, const float* __restrict__ Wb,
    const float* __restrict__ x, float* __restrict__ out)
{
  const int lane = threadIdx.x;
  const int b    = blockIdx.z;
  const int l0   = blockIdx.x * 16;
  const int obeg = blockIdx.y * 1024;          // 4 o-chunks of 1024
  const int row  = lane & 15;
  const int half = lane >> 4;
  const int kA   = half * 8;

  const __bf16* yrow = YB + ((size_t)b * NN + l0 + row) * NI + kA;
  const v16bf a0 = load_frag_bf(yrow);
  const v16bf a1 = load_frag_bf(yrow + 32);

  for (int o0 = obeg; o0 < obeg + 1024; o0 += 16) {
    const __bf16* wp = WwB + (size_t)(o0 + row) * NI + kA;
    const v16bf b0 = load_frag_bf(wp);
    const v16bf b1 = load_frag_bf(wp + 32);
    v8f acc = vzero8();
    acc = wmma_bf16(a0, b0, acc);
    acc = wmma_bf16(a1, b1, acc);

    const int o = o0 + row;                    // C-layout column
    const float bi = Wb[o];
#pragma unroll
    for (int v = 0; v < 8; ++v) {
      const int l = l0 + 8 * half + v;         // C-layout row
      acc[v] += bi + x[((size_t)b * NN + l) * NN + o];
    }
    float* op = out + ((size_t)b * NN + o) * NN + l0 + 8 * half;
    const float4* af = (const float4*)&acc;
    *(float4*)(op)     = af[0];
    *(float4*)(op + 4) = af[1];

    // pin: 13 reads (4 W frags + 8 residual + bias) ahead of the 2 WMMAs
    __builtin_amdgcn_sched_group_barrier(SG_VMEM_READ, 13, 0);
    __builtin_amdgcn_sched_group_barrier(SG_MFMA,       2, 0);
  }
}

// ---------------------------------------------------------------------------
extern "C" void kernel_launch(void* const* d_in, const int* in_sizes, int n_in,
                              void* d_out, int out_size, void* d_ws, size_t ws_size,
                              hipStream_t stream) {
  const float* x   = (const float*)d_in[0];
  const float* g_w = (const float*)d_in[1];
  const float* g_b = (const float*)d_in[2];
  const float* t_w = (const float*)d_in[3];
  const float* t_b = (const float*)d_in[4];
  const float* p_w = (const float*)d_in[5];
  const float* p_b = (const float*)d_in[6];
  const float* W_w = (const float*)d_in[7];
  const float* W_b = (const float*)d_in[8];
  float* out = (float*)d_out;

  const size_t S  = (size_t)NB * NN * NI;      // 1,048,576 panel elements
  const size_t WE = (size_t)NI * NN;           // 262,144 weight elements
  char* w = (char*)d_ws;
  float*  Pg   = (float*)w;   w += S * sizeof(float);
  float*  cmax = (float*)w;   w += (size_t)NB * NN * sizeof(float);
  float*  csum = (float*)w;   w += (size_t)NB * NN * sizeof(float);
  __bf16* PthB = (__bf16*)w;  w += S * sizeof(__bf16);
  __bf16* PphB = (__bf16*)w;  w += S * sizeof(__bf16);
  __bf16* gsTB = (__bf16*)w;  w += S * sizeof(__bf16);
  __bf16* YB   = (__bf16*)w;  w += S * sizeof(__bf16);
  __bf16* gwB  = (__bf16*)w;  w += WE * sizeof(__bf16);
  __bf16* twB  = (__bf16*)w;  w += WE * sizeof(__bf16);
  __bf16* pwB  = (__bf16*)w;  w += WE * sizeof(__bf16);
  __bf16* WwB  = (__bf16*)w;  w += WE * sizeof(__bf16);

  const int cvtBlocks = (int)(WE / 4 / 256);   // 256
  k_cvt<<<cvtBlocks, 256, 0, stream>>>(g_w, gwB);
  k_cvt<<<cvtBlocks, 256, 0, stream>>>(t_w, twB);
  k_cvt<<<cvtBlocks, 256, 0, stream>>>(p_w, pwB);
  k_cvt<<<cvtBlocks, 256, 0, stream>>>(W_w, WwB);

  dim3 gtiles(NN / 16, NB);                    // (256, 4)
  k_proj   <<<gtiles, 32, 0, stream>>>(x, gwB, g_b, twB, t_b, pwB, p_b,
                                       Pg, PthB, PphB);
  k_stats  <<<gtiles, 32, 0, stream>>>(PthB, PphB, cmax, csum);
  k_scale_g<<<dim3(NN / 256, NI, NB), 256, 0, stream>>>(Pg, csum, gsTB);
  k_attn   <<<gtiles, 32, 0, stream>>>(PthB, PphB, gsTB, cmax, YB);
  k_out    <<<dim3(NN / 16, 4, NB), 32, 0, stream>>>(YB, WwB, W_b, x, out);
}
